// RetrievalMemoryBank_80032420594095
// MI455X (gfx1250) — matrix operand order repaired and checked by reading, not compiled, not run on hardware
//
#include <hip/hip_runtime.h>
#include <hip/hip_bf16.h>
#include <math.h>

#define TOPK 50
#define TEMP 0.07f
#define DDIM 128
#define CDIM 256

typedef __bf16 bf16_t;
typedef __attribute__((ext_vector_type(16))) __bf16 v16bf;
typedef __attribute__((ext_vector_type(8)))  __bf16 v8bf;
typedef __attribute__((ext_vector_type(8)))  float  v8f;
typedef __attribute__((ext_vector_type(4)))  unsigned u32x4;
typedef __attribute__((ext_vector_type(4)))  int      i32x4;
typedef __attribute__((ext_vector_type(8)))  int      i32x8;

__device__ __forceinline__ bf16_t f2bf(float f) {
  union { float f; unsigned u; } v; v.f = f;
  unsigned r = v.u + 0x7FFFu + ((v.u >> 16) & 1u);   // round-to-nearest-even
  unsigned short h = (unsigned short)(r >> 16);
  bf16_t o; __builtin_memcpy(&o, &h, 2); return o;
}

__device__ __forceinline__ v8f wmma_bf16(v16bf a, v16bf b, v8f c) {
  return __builtin_amdgcn_wmma_f32_16x16x32_bf16(false, a, false, b, (short)0, c,
                                                 false, false);
}

// A fragment: 16x32 bf16 tile at (row 0, col k0), row-major, leading dim ld.
// Lane layout (ISA 7.12.2): lanes 0-15 = rows; per lane two contiguous 8-elem
// runs at K = k0+half*8 and K = k0+16+half*8  -> two 16B loads.
__device__ __forceinline__ v16bf load_a_frag(const bf16_t* p, int ld, int k0, int lane) {
  const int m = lane & 15, half = lane >> 4;
  const bf16_t* r = p + m * ld + k0 + half * 8;
  v8bf lo = *(const v8bf*)(r);
  v8bf hi = *(const v8bf*)(r + 16);
  return __builtin_shufflevector(lo, hi, 0, 1, 2, 3, 4, 5, 6, 7,
                                 8, 9, 10, 11, 12, 13, 14, 15);
}

// B fragment: element (kk, n) = p[(n0+n)*ld + kk], lane n needs two 8-elem
// runs at kk = k0+half*8 and k0+16+half*8 -> two 16B loads.
__device__ __forceinline__ v16bf load_b_frag(const bf16_t* p, int ld, int n0, int k0,
                                             int lane) {
  const int n = lane & 15, half = lane >> 4;
  const bf16_t* r = p + (size_t)(n0 + n) * ld + k0 + half * 8;
  v8bf lo = *(const v8bf*)(r);
  v8bf hi = *(const v8bf*)(r + 16);
  return __builtin_shufflevector(lo, hi, 0, 1, 2, 3, 4, 5, 6, 7,
                                 8, 9, 10, 11, 12, 13, 14, 15);
}

// ---------------------------------------------------------------------------
// TDM: async 2D tile load Global->LDS (rows x row_elems bf16, packed rows in
// memory, 272B padded rows in LDS via pad_interval=64DW, pad_amount=4DW).
// D# per CDNA5 ISA ch.8: group0 = {count, lds_addr, global_addr, type=2},
// group1 = {data_size=2B, pads, tensor dims, tile dims, strides}.
__device__ __forceinline__ void tdm_load_2d_bf16(unsigned lds_addr, const void* gptr,
                                                 unsigned rows, unsigned row_elems) {
  unsigned long long ga = (unsigned long long)(size_t)gptr;
  u32x4 g0;
  g0[0] = 1u;                                            // count=1, user mode
  g0[1] = lds_addr;                                      // LDS byte address
  g0[2] = (unsigned)(ga & 0xffffffffu);                  // global_addr lo
  g0[3] = (unsigned)((ga >> 32) & 0x1ffffffu) | (2u << 30);  // addr hi | type=2
  i32x8 g1;
  g1[0] = (int)((1u << 16) | (1u << 20) | (5u << 22) | (3u << 25));
  //            data_size=2B  pad_en      every 64 DW    +4 DW pad
  g1[1] = (int)(row_elems << 16);            // tensor_dim0[15:0]
  g1[2] = (int)((row_elems >> 16) | (rows << 16));   // dim0 hi | tensor_dim1 lo
  g1[3] = (int)((rows >> 16) | (row_elems << 16));   // dim1 hi | tile_dim0
  g1[4] = (int)rows;                         // tile_dim1 (tile_dim2 = 0 -> 2D)
  g1[5] = (int)row_elems;                    // tensor_dim0_stride lo
  g1[6] = 0;
  g1[7] = 0;
  i32x4 z4 = {};
#if defined(__clang_major__) && __clang_major__ >= 23
  i32x8 z8 = {};
  __builtin_amdgcn_tensor_load_to_lds(g0, g1, z4, z4, z8, 0);
#else
  __builtin_amdgcn_tensor_load_to_lds(g0, g1, z4, z4, 0);
#endif
}

// ---------------------------------------------------------------------------
// fp32 [K][N] -> bf16 transposed [N][K] (so WMMA B-fragments are contiguous)
__global__ void cvt_transpose_bf16_kernel(const float* __restrict__ in,
                                          bf16_t* __restrict__ out, int K, int N) {
  int i = blockIdx.x * blockDim.x + threadIdx.x;
  if (i < K * N) {
    int k = i / N, n = i % N;
    out[(size_t)n * K + k] = f2bf(in[i]);
  }
}

// ---------------------------------------------------------------------------
// out[r,:] = bf16( l2norm( X[r,:] @ W + bias ) ), 16 rows per block.
// WT is W transposed [N][K] bf16.  Padded rows written as zeros.
__global__ void proj_norm_kernel(const float* __restrict__ X,
                                 const bf16_t* __restrict__ WT,
                                 const float* __restrict__ bias,
                                 bf16_t* __restrict__ out, int nrows) {
  __shared__ bf16_t xs[16][136];
  __shared__ float  os[16][136];
  __shared__ float  rinv[16];
  const int tid = threadIdx.x, lane = tid & 31, wave = tid >> 5;
  const int rowbase = blockIdx.x * 16;

  for (int e = tid; e < 16 * DDIM; e += 128) {
    int r = e >> 7, c = e & 127;
    float v = (rowbase + r < nrows) ? X[(size_t)(rowbase + r) * DDIM + c] : 0.f;
    xs[r][c] = f2bf(v);
  }
  __syncthreads();

  for (int t = 0; t < 2; ++t) {                 // each wave: 2 col-tiles of 16
    const int ct = wave * 2 + t;
    v8f acc = {};
#pragma unroll
    for (int ks = 0; ks < 4; ++ks) {            // K = 128 in 4 x K32
      v16bf a = load_a_frag(&xs[0][0], 136, ks * 32, lane);
      v16bf b = load_b_frag(WT, DDIM, ct * 16, ks * 32, lane);
      acc = wmma_bf16(a, b, acc);
    }
    const int n = ct * 16 + (lane & 15);
    const int half = lane >> 4;
    float bn = bias[n];
#pragma unroll
    for (int r = 0; r < 8; ++r) os[half * 8 + r][n] = acc[r] + bn;
  }
  __syncthreads();
  if (tid < 16) {
    float s = 0.f;
    for (int c = 0; c < DDIM; ++c) { float v = os[tid][c]; s += v * v; }
    rinv[tid] = 1.f / fmaxf(sqrtf(s), 1e-12f);
  }
  __syncthreads();
  for (int e = tid * 8; e < 16 * DDIM; e += 128 * 8) {
    int r = e >> 7, c = e & 127;
    bool ok = (rowbase + r < nrows);
    v8bf o;
#pragma unroll
    for (int i = 0; i < 8; ++i) o[i] = f2bf(ok ? os[r][c + i] * rinv[r] : 0.f);
    *(v8bf*)&out[(size_t)(rowbase + r) * DDIM + c] = o;
  }
}

// ---------------------------------------------------------------------------
// sims = q @ k^T fused with session masking + per-row top-50.
// 16 query rows per block, 8 waves.  Key chunks (128 rows x 128 bf16) are
// double-buffered into LDS by the Tensor Data Mover (prefetch next chunk
// while WMMAs consume the current one); scan uses wave32 ballot filtering.
__global__ void sims_topk_kernel(const bf16_t* __restrict__ qbf,
                                 const bf16_t* __restrict__ kbf,
                                 const int* __restrict__ session_ids,
                                 const int* __restrict__ session_queue,
                                 float* __restrict__ topv, int* __restrict__ topi,
                                 int B, int M, int Mpad) {
  __shared__ bf16_t qs[16][136];
  __shared__ bf16_t kbuf[2][128][136];   // 272B rows == TDM pad scheme
  __shared__ float  stile[16][132];
  __shared__ int    sq[128];
  __shared__ float  lv[16][TOPK];
  __shared__ int    li[16][TOPK];
  __shared__ int    sid_s[16];
  const int tid = threadIdx.x, lane = tid & 31, wave = tid >> 5;
  const int rowbase = blockIdx.x * 16;

  {                                      // stage 16x128 q tile, 16B per thread
    int r = tid >> 4, c = (tid & 15) * 8;
    *(v8bf*)&qs[r][c] = *(const v8bf*)&qbf[(size_t)(rowbase + r) * DDIM + c];
  }
  if (tid < 16)
    sid_s[tid] = (rowbase + tid < B) ? session_ids[rowbase + tid] : (int)0x80000000;
  for (int e = tid; e < 16 * TOPK; e += 256) {
    lv[e / TOPK][e % TOPK] = -INFINITY;
    li[e / TOPK][e % TOPK] = -1;
  }

  const int nch = Mpad / 128;
  if (wave == 0)
    tdm_load_2d_bf16((unsigned)(size_t)&kbuf[0][0][0], kbf, 128, DDIM);

  for (int ci = 0; ci < nch; ++ci) {
    const int base = ci * 128;
    const bf16_t* kcur = &kbuf[ci & 1][0][0];
    bf16_t* knxt = (bf16_t*)&kbuf[(ci + 1) & 1][0][0];
    if (tid < 128) {
      int c = base + tid;
      sq[tid] = (c < M) ? session_queue[c] : 0x7fffffff;
    }
    if (wave == 0) {
      if (ci + 1 < nch) {
        tdm_load_2d_bf16((unsigned)(size_t)knxt,
                         kbf + (size_t)(base + 128) * DDIM, 128, DDIM);
        __builtin_amdgcn_s_wait_tensorcnt(1);   // current chunk resident
      } else {
        __builtin_amdgcn_s_wait_tensorcnt(0);
      }
    }
    __syncthreads();

    // 16x16 tile at columns [base + wave*16, +16), K=128 in 4 x K32
    v8f acc = {};
#pragma unroll
    for (int ks = 0; ks < 4; ++ks) {
      v16bf a = load_a_frag(&qs[0][0], 136, ks * 32, lane);
      v16bf b = load_b_frag(kcur, 136, wave * 16, ks * 32, lane);
      acc = wmma_bf16(a, b, acc);
    }
    {
      const int n = wave * 16 + (lane & 15);
      const int half = lane >> 4;
#pragma unroll
      for (int r = 0; r < 8; ++r) stile[half * 8 + r][n] = acc[r];
    }
    __syncthreads();

    // each wave scans 2 rows of the 16x128 tile
#pragma unroll
    for (int rr = 0; rr < 2; ++rr) {
      const int r = wave * 2 + rr;
      const int sid = sid_s[r];
      for (int g = 0; g < 4; ++g) {
        const int cloc = g * 32 + lane;
        const int col = base + cloc;
        const float s = stile[r][cloc];
        const float thr = lv[r][TOPK - 1];
        bool ok = (col < M) && (sq[cloc] != sid) && (s > thr);
        unsigned long long mask = __ballot(ok);
        while (mask) {
          int j = (int)__builtin_ctzll(mask);
          mask &= mask - 1;
          float v = __shfl(s, j, 32);
          int   c2 = __shfl(col, j, 32);
          if (lane == 0 && v > lv[r][TOPK - 1]) {   // re-check vs fresh min
            int p = TOPK - 1;
            while (p > 0 && lv[r][p - 1] < v) {
              lv[r][p] = lv[r][p - 1];
              li[r][p] = li[r][p - 1];
              --p;
            }
            lv[r][p] = v;
            li[r][p] = c2;
          }
        }
      }
    }
    __syncthreads();
  }

  for (int e = tid; e < 16 * TOPK; e += 256) {
    int r = e / TOPK, j = e % TOPK;
    int row = rowbase + r;
    if (row < B) {
      topv[(size_t)row * TOPK + j] = lv[r][j];
      topi[(size_t)row * TOPK + j] = li[r][j];
    }
  }
}

// ---------------------------------------------------------------------------
// Per-query: softmax over top-50, fp32 context gather, neighbor MLP
// gelu([feat,tgt] @ W1 + b1) @ W2 + b2 as WMMA bf16, attn-weighted summary.
__global__ void finalize_kernel(const float* __restrict__ topv,
                                const int* __restrict__ topi,
                                const float* __restrict__ feature_queue,
                                const float* __restrict__ item_emb,
                                const int* __restrict__ target_queue,
                                const bf16_t* __restrict__ W1T,  // [C][2D]
                                const float* __restrict__ b1,
                                const bf16_t* __restrict__ W2T,  // [C][C]
                                const float* __restrict__ b2,
                                const float* __restrict__ fb_ctx,
                                const float* __restrict__ fb_sum,
                                float* __restrict__ out_ctx,
                                float* __restrict__ out_sum,
                                float* __restrict__ out_used) {
  __shared__ float  attn[64];
  __shared__ int    nidx[64];
  __shared__ int    tgi[64];
  __shared__ bf16_t Hin[64][264];     // [feat | tgt_emb], rows 50..63 zero
  __shared__ bf16_t Hmt[16][264];     // one row-tile of gelu(Hin@W1+b1)
  __shared__ float  ssum[CDIM];
  __shared__ int    usedflag;
  const int tid = threadIdx.x, lane = tid & 31, wave = tid >> 5;
  const int b = blockIdx.x;

  if (tid < 64) { attn[tid] = -INFINITY; nidx[tid] = -1; tgi[tid] = 0; }
  if (tid == 0) usedflag = 0;
  __syncthreads();
  if (tid < TOPK) {
    float v = topv[(size_t)b * TOPK + tid];
    int   i = topi[(size_t)b * TOPK + tid];
    bool valid = (v > -1e30f) && (i >= 0);
    attn[tid] = valid ? v : -INFINITY;
    nidx[tid] = valid ? i : -1;
    if (valid) { int t = target_queue[i]; tgi[tid] = t > 0 ? t : 0; }
  }
  __syncthreads();
  if (tid == 0) {                          // softmax over valid entries
    float mx = -INFINITY;
    for (int j = 0; j < TOPK; ++j)
      if (nidx[j] >= 0 && attn[j] > mx) mx = attn[j];
    if (mx > -1e30f) {
      float s = 0.f;
      for (int j = 0; j < TOPK; ++j) {
        if (nidx[j] >= 0) { float e = __expf((attn[j] - mx) / TEMP); attn[j] = e; s += e; }
        else attn[j] = 0.f;
      }
      float inv = 1.f / fmaxf(s, 1e-12f);
      for (int j = 0; j < TOPK; ++j) attn[j] *= inv;
      usedflag = 1;
    } else {
      for (int j = 0; j < TOPK; ++j) attn[j] = 0.f;
    }
    for (int j = TOPK; j < 64; ++j) attn[j] = 0.f;
  }
  __syncthreads();
  if (!usedflag) {
    if (tid < DDIM) out_ctx[(size_t)b * DDIM + tid] = fb_ctx[tid];
    if (tid < CDIM) out_sum[(size_t)b * CDIM + tid] = fb_sum[tid];
    if (tid == 0) out_used[b] = 0.f;
    return;
  }

  for (int e = tid; e < 64 * (2 * DDIM); e += 256) {   // build Hin (bf16)
    int r = e >> 8, c = e & 255;
    float v = 0.f;
    if (r < TOPK && nidx[r] >= 0)
      v = (c < DDIM) ? feature_queue[(size_t)nidx[r] * DDIM + c]
                     : item_emb[(size_t)tgi[r] * DDIM + (c - DDIM)];
    Hin[r][c] = f2bf(v);
  }
  if (tid < CDIM) ssum[tid] = 0.f;
  if (tid < DDIM) {                                    // fp32 context gather
    float acc = 0.f;
    for (int j = 0; j < TOPK; ++j)
      if (nidx[j] >= 0) acc += attn[j] * feature_queue[(size_t)nidx[j] * DDIM + tid];
    out_ctx[(size_t)b * DDIM + tid] = acc;
  }
  __syncthreads();

  for (int rt = 0; rt < 4; ++rt) {          // pipeline row-tiles through MLP
    // GEMM1 + GELU: Hmt = gelu(Hin[rt] @ W1 + b1); 2 col-tiles per wave
    for (int t = 0; t < 2; ++t) {
      const int ct = wave * 2 + t;
      v8f acc = {};
#pragma unroll
      for (int ks = 0; ks < 8; ++ks) {
        v16bf a = load_a_frag(&Hin[rt * 16][0], 264, ks * 32, lane);
        v16bf bb = load_b_frag(W1T, 2 * DDIM, ct * 16, ks * 32, lane);
        acc = wmma_bf16(a, bb, acc);
      }
      const int n = ct * 16 + (lane & 15);
      const int half = lane >> 4;
      float bn = b1[n];
#pragma unroll
      for (int r = 0; r < 8; ++r) {
        float x = acc[r] + bn;
        float g = 0.5f * x * (1.f + erff(x * 0.70710678f));   // exact gelu
        Hmt[half * 8 + r][n] = f2bf(g);
      }
    }
    __syncthreads();
    // GEMM2 + attn-weighted accumulation into ssum
    for (int t = 0; t < 2; ++t) {
      const int ct = wave * 2 + t;
      v8f acc = {};
#pragma unroll
      for (int ks = 0; ks < 8; ++ks) {
        v16bf a = load_a_frag(&Hmt[0][0], 264, ks * 32, lane);
        v16bf bb = load_b_frag(W2T, CDIM, ct * 16, ks * 32, lane);
        acc = wmma_bf16(a, bb, acc);
      }
      const int n = ct * 16 + (lane & 15);
      const int half = lane >> 4;
      float bn = b2[n];
#pragma unroll
      for (int r = 0; r < 8; ++r) {
        float w = attn[rt * 16 + half * 8 + r];
        if (w != 0.f) atomicAdd(&ssum[n], w * (acc[r] + bn));
      }
    }
    __syncthreads();
  }
  if (tid < CDIM) out_sum[(size_t)b * CDIM + tid] = ssum[tid];
  if (tid == 0) out_used[b] = 1.f;
}

// ---------------------------------------------------------------------------
extern "C" void kernel_launch(void* const* d_in, const int* in_sizes, int n_in,
                              void* d_out, int out_size, void* d_ws, size_t ws_size,
                              hipStream_t stream) {
  (void)n_in; (void)out_size; (void)ws_size;
  const float* current_repr  = (const float*)d_in[0];
  const int*   session_ids   = (const int*)d_in[1];
  const float* item_emb      = (const float*)d_in[2];
  const float* feature_queue = (const float*)d_in[3];
  const int*   session_queue = (const int*)d_in[4];
  const int*   target_queue  = (const int*)d_in[5];
  const float* Wq = (const float*)d_in[6];
  const float* bq = (const float*)d_in[7];
  const float* Wk = (const float*)d_in[8];
  const float* bk = (const float*)d_in[9];
  const float* W1 = (const float*)d_in[10];
  const float* b1 = (const float*)d_in[11];
  const float* W2 = (const float*)d_in[12];
  const float* b2 = (const float*)d_in[13];
  const float* fb_ctx = (const float*)d_in[14];
  const float* fb_sum = (const float*)d_in[15];

  const int B = in_sizes[1];
  const int M = in_sizes[4];
  const int Mpad = (M + 127) & ~127;
  const int Bpad = (B + 15) & ~15;

  char* w = (char*)d_ws;
  bf16_t* k_bf  = (bf16_t*)w; w += (size_t)Mpad * DDIM * sizeof(bf16_t);
  bf16_t* q_bf  = (bf16_t*)w; w += (size_t)Bpad * DDIM * sizeof(bf16_t);
  bf16_t* WqT   = (bf16_t*)w; w += (size_t)DDIM * DDIM * sizeof(bf16_t);
  bf16_t* WkT   = (bf16_t*)w; w += (size_t)DDIM * DDIM * sizeof(bf16_t);
  bf16_t* W1T   = (bf16_t*)w; w += (size_t)(2 * DDIM) * CDIM * sizeof(bf16_t);
  bf16_t* W2T   = (bf16_t*)w; w += (size_t)CDIM * CDIM * sizeof(bf16_t);
  float*  topv  = (float*)w;  w += (size_t)B * TOPK * sizeof(float);
  int*    topi  = (int*)w;    w += (size_t)B * TOPK * sizeof(int);

  cvt_transpose_bf16_kernel<<<(DDIM * DDIM + 255) / 256, 256, 0, stream>>>(Wq, WqT, DDIM, DDIM);
  cvt_transpose_bf16_kernel<<<(DDIM * DDIM + 255) / 256, 256, 0, stream>>>(Wk, WkT, DDIM, DDIM);
  cvt_transpose_bf16_kernel<<<(2 * DDIM * CDIM + 255) / 256, 256, 0, stream>>>(W1, W1T, 2 * DDIM, CDIM);
  cvt_transpose_bf16_kernel<<<(CDIM * CDIM + 255) / 256, 256, 0, stream>>>(W2, W2T, CDIM, CDIM);

  proj_norm_kernel<<<Bpad / 16, 128, 0, stream>>>(current_repr, WqT, bq, q_bf, B);
  proj_norm_kernel<<<Mpad / 16, 128, 0, stream>>>(feature_queue, WkT, bk, k_bf, M);

  sims_topk_kernel<<<Bpad / 16, 256, 0, stream>>>(q_bf, k_bf, session_ids, session_queue,
                                                  topv, topi, B, M, Mpad);

  float* out_ctx  = (float*)d_out;
  float* out_sum  = out_ctx + (size_t)B * DDIM;
  float* out_used = out_sum + (size_t)B * CDIM;
  finalize_kernel<<<B, 256, 0, stream>>>(topv, topi, feature_queue, item_emb, target_queue,
                                         W1T, b1, W2T, b2, fb_ctx, fb_sum,
                                         out_ctx, out_sum, out_used);
}
